// Quantize_15135464751618
// MI455X (gfx1250) — matrix operand order repaired and checked by reading, not compiled, not run on hardware
//
#include <hip/hip_runtime.h>

// ---------------------------------------------------------------------------
// Fused vector-quantize + gumbel-softmax embedding for MI455X (gfx1250).
//   N=131072 rows, D=64, K=1024 codes.
// v3: codebook (f16 row-major + f16 transposed + norms, 266 KB) is DMA'd into
// LDS once per workgroup via the CDNA5 Tensor Data Mover (tensor_load_to_lds,
// s_wait_tensorcnt — 6-arg builtin form on this toolchain), then all WMMA
// B-fragments are fed from ds_load_b128.
// One wave32 per 16-row tile; both GEMMs on v_wmma_f32_16x16x32_f16; the
// [N,K] logits never leave registers (two-pass online softmax).
// ---------------------------------------------------------------------------

typedef _Float16 v16h __attribute__((ext_vector_type(16)));
typedef _Float16 h8   __attribute__((ext_vector_type(8)));
typedef float    v8f  __attribute__((ext_vector_type(8)));
typedef unsigned int u32x4 __attribute__((ext_vector_type(4)));
typedef int          i32x4 __attribute__((ext_vector_type(4)));
typedef int          i32x8 __attribute__((ext_vector_type(8)));

#define NROWS 131072
#define DDIM  64
#define KCODE 1024

// LDS image of the workspace codebook region (copied verbatim by the TDM),
// plus per-wave weight-tile staging. 274,432 bytes < 320 KB WGP LDS.
struct __align__(16) LdsBuf {
    _Float16 ch[KCODE * DDIM];    // [1024][64] f16 row-major   (131072 B)
    _Float16 cth[DDIM * KCODE];   // [64][1024] f16 transposed  (131072 B)
    float    cnorm[KCODE];        // ||c_k||^2                  (  4096 B)
    _Float16 wst[8][16][32];      // per-wave GEMM2 A staging   (  8192 B)
};
#define WS_CODE_BYTES (KCODE * DDIM * 2 * 2 + KCODE * 4)   // 266240

// Deterministic gumbel noise per (row, code): integer hash -> U(0,1) -> gumbel.
__device__ __forceinline__ float gumbel_noise(unsigned int idx) {
    unsigned int h = idx;
    h ^= h >> 16; h *= 0x7feb352du;
    h ^= h >> 15; h *= 0x846ca68bu;
    h ^= h >> 16;
    float u = (float)(h >> 8) * (1.0f / 16777216.0f) + 2.9802322e-8f;
    return -__logf(-__logf(u));
}

// Prep: codebook f32 -> f16 row-major (ch), f16 transposed (cth), and norms,
// written contiguously into the workspace in LdsBuf order.
__global__ void vq_prep_codebook(const float* __restrict__ cb,
                                 _Float16* __restrict__ ch,
                                 _Float16* __restrict__ cth,
                                 float* __restrict__ cnorm) {
    int k = blockIdx.x * blockDim.x + threadIdx.x;
    if (k >= KCODE) return;
    float acc = 0.0f;
    #pragma unroll 4
    for (int d = 0; d < DDIM; ++d) {
        float v = cb[k * DDIM + d];
        acc += v * v;
        _Float16 hv = (_Float16)v;
        ch[k * DDIM + d]   = hv;
        cth[d * KCODE + k] = hv;
    }
    cnorm[k] = acc;
}

__global__ void __launch_bounds__(256)
vq_fused(const float* __restrict__ x,
         const float* __restrict__ temp,
         const void* __restrict__ ws_code,   // ch ++ cth ++ cnorm, contiguous
         float* __restrict__ out_emb,
         float* __restrict__ out_ids) {
    __shared__ LdsBuf lds;

    const int lane = threadIdx.x & 31;
    const int wave = threadIdx.x >> 5;
    const int hl   = lane & 15;    // WMMA N / M index within half
    const int hi   = lane >> 4;    // which 16-lane half
    const int row0 = (blockIdx.x * 8 + wave) * 16;   // 16 rows per wave

    // ---- TDM: DMA the whole codebook region into LDS (one op per block) ----
    if (wave == 0) {
        const unsigned long long ga = (unsigned long long)(uintptr_t)ws_code;
        const unsigned lo = (unsigned)(uintptr_t)&lds;   // LDS byte offset
        const unsigned elems = WS_CODE_BYTES / 8;        // 33280 x 8B elements

        u32x4 g0;
        g0[0] = 1u;                                      // count=1 (valid D#)
        g0[1] = lo;                                      // lds_addr
        g0[2] = (unsigned)ga;                            // global_addr[31:0]
        g0[3] = (unsigned)(ga >> 32) | (2u << 30);       // addr[56:32] | type=2

        i32x8 g1;
        g1[0] = (int)(3u << 16);                         // data_size=3 (8B), mask=0
        g1[1] = (int)((elems & 0xFFFFu) << 16);          // tensor_dim0[15:0]
        g1[2] = (int)((elems >> 16) | (1u << 16));       // tensor_dim0[31:16] | tensor_dim1=1
        g1[3] = (int)((elems & 0xFFFFu) << 16);          // tile_dim0
        g1[4] = 1;                                       // tile_dim1=1, tile_dim2=0
        g1[5] = (int)elems;                              // tensor_dim0_stride[31:0]
        g1[6] = 0;
        g1[7] = 0;

        i32x4 z4 = {0, 0, 0, 0};
        i32x8 z8 = {0, 0, 0, 0, 0, 0, 0, 0};
        __builtin_amdgcn_tensor_load_to_lds(g0, g1, z4, z4, z8, 0);
        __builtin_amdgcn_s_wait_tensorcnt(0);
    }
    __syncthreads();

    const float invT = 1.0f / temp[0];

    // ---- load x rows as two f16 A-fragments (K = D dims 0..31 and 32..63) ----
    v16h a0, a1;
    {
        const float* xr = x + (size_t)(row0 + hl) * DDIM;
        #pragma unroll
        for (int j = 0; j < 8; ++j) {
            a0[j]     = (_Float16)xr[hi * 16 + j];
            a0[8 + j] = (_Float16)xr[hi * 16 + 8 + j];
            a1[j]     = (_Float16)xr[32 + hi * 16 + j];
            a1[8 + j] = (_Float16)xr[32 + hi * 16 + 8 + j];
        }
    }

    // =========================== pass 1: max / argmax ========================
    float m[8], bestv[8];
    int   besti[8];
    #pragma unroll
    for (int r = 0; r < 8; ++r) { m[r] = -3.4e38f; bestv[r] = -3.4e38f; besti[r] = 0; }

    for (int t = 0; t < 64; ++t) {
        const int code = t * 16 + hl;
        const _Float16* cr = &lds.ch[(unsigned)code * DDIM];
        h8 p0 = *(const h8*)(cr + hi * 16);
        h8 p1 = *(const h8*)(cr + hi * 16 + 8);
        h8 p2 = *(const h8*)(cr + 32 + hi * 16);
        h8 p3 = *(const h8*)(cr + 32 + hi * 16 + 8);
        v16h b0, b1;
        #pragma unroll
        for (int j = 0; j < 8; ++j) {
            b0[j] = p0[j]; b0[8 + j] = p1[j];
            b1[j] = p2[j]; b1[8 + j] = p3[j];
        }
        v8f acc = {};
        acc = __builtin_amdgcn_wmma_f32_16x16x32_f16(false, a0, false, b0,
                                                     (short)0, acc, false, false);
        acc = __builtin_amdgcn_wmma_f32_16x16x32_f16(false, a1, false, b1,
                                                     (short)0, acc, false, false);
        const float cn = lds.cnorm[code];
        #pragma unroll
        for (int r = 0; r < 8; ++r) {
            const int grow = row0 + r + 8 * hi;
            float clean = 2.0f * acc[r] - cn;      // per-row ||x||^2 cancels
            float noisy = (clean + gumbel_noise((unsigned)grow * KCODE + code)) * invT;
            m[r] = fmaxf(m[r], noisy);
            if (clean > bestv[r]) { bestv[r] = clean; besti[r] = code; }
        }
    }
    #pragma unroll
    for (int r = 0; r < 8; ++r) {
        #pragma unroll
        for (int off = 1; off < 16; off <<= 1) {
            float ov = __shfl_xor(bestv[r], off, 32);
            int   oi = __shfl_xor(besti[r], off, 32);
            float om = __shfl_xor(m[r],     off, 32);
            if (ov > bestv[r] || (ov == bestv[r] && oi < besti[r])) {
                bestv[r] = ov; besti[r] = oi;
            }
            m[r] = fmaxf(m[r], om);
        }
    }

    // ================== pass 2: exp, sum, emb = W @ codebook =================
    v8f e[4] = {v8f{}, v8f{}, v8f{}, v8f{}};
    float s[8];
    #pragma unroll
    for (int r = 0; r < 8; ++r) s[r] = 0.0f;

    for (int kb = 0; kb < 32; ++kb) {          // 32 codes per iteration
        #pragma unroll
        for (int sub = 0; sub < 2; ++sub) {
            const int code = (kb * 2 + sub) * 16 + hl;
            const _Float16* cr = &lds.ch[(unsigned)code * DDIM];
            h8 p0 = *(const h8*)(cr + hi * 16);
            h8 p1 = *(const h8*)(cr + hi * 16 + 8);
            h8 p2 = *(const h8*)(cr + 32 + hi * 16);
            h8 p3 = *(const h8*)(cr + 32 + hi * 16 + 8);
            v16h b0, b1;
            #pragma unroll
            for (int j = 0; j < 8; ++j) {
                b0[j] = p0[j]; b0[8 + j] = p1[j];
                b1[j] = p2[j]; b1[8 + j] = p3[j];
            }
            v8f acc = {};
            acc = __builtin_amdgcn_wmma_f32_16x16x32_f16(false, a0, false, b0,
                                                         (short)0, acc, false, false);
            acc = __builtin_amdgcn_wmma_f32_16x16x32_f16(false, a1, false, b1,
                                                         (short)0, acc, false, false);
            const float cn = lds.cnorm[code];
            #pragma unroll
            for (int r = 0; r < 8; ++r) {
                const int grow = row0 + r + 8 * hi;
                float clean = 2.0f * acc[r] - cn;
                float noisy = (clean + gumbel_noise((unsigned)grow * KCODE + code)) * invT;
                float w = __expf(noisy - m[r]);
                s[r] += w;
                lds.wst[wave][r + 8 * hi][sub * 16 + hl] = (_Float16)w;
            }
        }
        __syncthreads();
        v16h aw;
        {
            const _Float16* wr = &lds.wst[wave][hl][0];
            h8 q0 = *(const h8*)(wr + hi * 16);
            h8 q1 = *(const h8*)(wr + hi * 16 + 8);
            #pragma unroll
            for (int j = 0; j < 8; ++j) { aw[j] = q0[j]; aw[8 + j] = q1[j]; }
        }
        __syncthreads();
        #pragma unroll
        for (int nt = 0; nt < 4; ++nt) {
            const _Float16* cc = &lds.cth[(unsigned)(nt * 16 + hl) * KCODE + kb * 32];
            h8 r0 = *(const h8*)(cc + hi * 16);
            h8 r1 = *(const h8*)(cc + hi * 16 + 8);
            v16h bw;
            #pragma unroll
            for (int j = 0; j < 8; ++j) { bw[j] = r0[j]; bw[8 + j] = r1[j]; }
            e[nt] = __builtin_amdgcn_wmma_f32_16x16x32_f16(false, aw, false, bw,
                                                           (short)0, e[nt], false, false);
        }
    }

    #pragma unroll
    for (int r = 0; r < 8; ++r) {
        #pragma unroll
        for (int off = 1; off < 16; off <<= 1)
            s[r] += __shfl_xor(s[r], off, 32);
    }

    // ------------------------------- write out -------------------------------
    #pragma unroll
    for (int r = 0; r < 8; ++r) {
        const int grow = row0 + r + 8 * hi;
        const float inv = 1.0f / s[r];
        #pragma unroll
        for (int nt = 0; nt < 4; ++nt)
            out_emb[(size_t)grow * DDIM + nt * 16 + hl] = e[nt][r] * inv;
        if (hl == 0)
            out_ids[grow] = (float)besti[r];
    }
}

extern "C" void kernel_launch(void* const* d_in, const int* in_sizes, int n_in,
                              void* d_out, int out_size, void* d_ws, size_t ws_size,
                              hipStream_t stream) {
    const float* x    = (const float*)d_in[0];
    const float* cb   = (const float*)d_in[1];
    const float* temp = (const float*)d_in[2];
    float* out = (float*)d_out;

    // workspace layout mirrors LdsBuf (ch ++ cth ++ cnorm), 266,240 bytes
    char* ws = (char*)d_ws;
    _Float16* ch    = (_Float16*)(ws);
    _Float16* cth   = (_Float16*)(ws + KCODE * DDIM * 2);
    float*    cnorm = (float*)   (ws + KCODE * DDIM * 2 * 2);

    vq_prep_codebook<<<4, 256, 0, stream>>>(cb, ch, cth, cnorm);

    // 8 waves/block, 16 rows per wave -> 128 rows/block, 1024 blocks
    vq_fused<<<NROWS / 128, 256, 0, stream>>>(
        x, temp, (const void*)ws,
        out,                         // emb [N,64]
        out + (size_t)NROWS * DDIM); // ids [N]
}
